// MultiHeadAttention_9457517986327
// MI455X (gfx1250) — matrix operand order repaired and checked by reading, not compiled, not run on hardware
//
#include <hip/hip_runtime.h>

// ---------------------------------------------------------------------------
// MultiHeadAttention (softmax over QUERY axis) for MI455X / gfx1250, wave32.
// All GEMMs run on V_WMMA_F32_16X16X32_BF16 (fp32 accumulate).
// Pipeline (all on `stream`):
//   1. fp32 -> bf16 converts (X) and convert+transpose (W, Wo)
//   2. Q/K/V projections: bf16 WMMA GEMM, bias epilogue       -> Qh/Kh/Vh bf16
//   3. scores = Q K^T / sqrt(dk): bf16 WMMA GEMM, fp32 out    -> scores fp32
//   4. per-KEY-column online softmax stats (max m[t], 1/Z[t])
//   5. heads = exp(S - m[t]) * (V[t]/Z[t]): fused exp in A-tile staging,
//      WMMA GEMM, bf16 out directly in concat layout [s, h*dk+k]
//   6. out = concat @ Wo + bo: WMMA GEMM, fp32 -> d_out
// GEMM staging uses GLOBAL_LOAD_ASYNC_TO_LDS_B128 + double-buffered LDS when
// the toolchain exposes the builtin (ASYNCcnt-tracked, overlaps WMMA).
// Workspace requirement: ~345 MB (11 x 8MB bf16 buffers + 256MB fp32 scores).
// ---------------------------------------------------------------------------

typedef __bf16 bf16;
typedef __attribute__((ext_vector_type(8)))  __bf16 v8bf;
typedef __attribute__((ext_vector_type(16))) __bf16 v16bf;
typedef __attribute__((ext_vector_type(8)))  float  v8f;
typedef __attribute__((ext_vector_type(4)))  float  v4f;
typedef int v4i __attribute__((__vector_size__(4 * sizeof(int))));

#define DEV __device__ __forceinline__

static const int SEQ = 2048;
static const int DM  = 2048;
static const int NH  = 16;
static const int DK  = 128;

// ------------------------- tiling config -----------------------------------
#define TM 128
#define TN 128
#define TK 64
#define LDSS (TK + 8)   // bf16 elements per LDS row (pad vs bank conflicts)

// ----------------- async global->LDS support (gfx1250) ----------------------
#if defined(__has_builtin)
#  if __has_builtin(__builtin_amdgcn_global_load_async_to_lds_b128)
#    define USE_ASYNC 1
#  endif
#endif
#ifndef USE_ASYNC
#  define USE_ASYNC 0
#endif

#if USE_ASYNC
#  if __has_builtin(__builtin_amdgcn_s_wait_asynccnt)
#    define WAIT_ASYNC(n) __builtin_amdgcn_s_wait_asynccnt(n)
#  else
#    define WAIT_ASYNC(n) asm volatile("s_wait_asynccnt %0" ::"n"(n) : "memory")
#  endif
// Builtin signature (from hipcc diagnostic): param0 is addrspace(1) v4i*,
// i.e. typed by the 128-bit payload; LDS side is the addrspace(3) analog.
DEV void async_cp_b128(const void* g, void* l) {
  __builtin_amdgcn_global_load_async_to_lds_b128(
      (__attribute__((address_space(1))) v4i*)(void*)g,
      (__attribute__((address_space(3))) v4i*)l, 0, 0);
}
#endif

// ---------------------- elementwise converts --------------------------------
__global__ void cvt_bf16_kernel(const float* __restrict__ in,
                                bf16* __restrict__ out, int n) {
  int i = (blockIdx.x * blockDim.x + threadIdx.x) * 4;
  if (i + 3 < n) {
    v4f x = *(const v4f*)(in + i);
    out[i + 0] = (bf16)x.x;
    out[i + 1] = (bf16)x.y;
    out[i + 2] = (bf16)x.z;
    out[i + 3] = (bf16)x.w;
  } else {
    for (; i < n; ++i) out[i] = (bf16)in[i];
  }
}

// out[b][n][d] = (bf16) in[b][d][n]   (store B^T so GEMM B-tiles are contiguous)
__global__ void cvt_transpose_kernel(const float* __restrict__ in,
                                     bf16* __restrict__ out,
                                     int D, int N, int batch) {
  size_t idx = (size_t)blockIdx.x * blockDim.x + threadIdx.x;
  size_t total = (size_t)batch * D * N;
  if (idx >= total) return;
  int d = (int)(idx % D);
  int n = (int)((idx / D) % N);
  int b = (int)(idx / ((size_t)D * N));
  out[idx] = (bf16)in[((size_t)b * D + d) * N + n];
}

// --------------------- WMMA fragment loaders (LDS) --------------------------
// A (16x32 bf16, MxK): lane L<16 holds row M=L, K = {k0..k0+7, k0+16..k0+23};
//                      lane L>=16 same row, K = {k0+8..k0+15, k0+24..k0+31}.
DEV v16bf load_frag_a(const bf16* s, int m0, int k0, int lane) {
  const bf16* p = s + (m0 + (lane & 15)) * LDSS + k0 + ((lane >> 4) << 3);
  v8bf lo = *(const v8bf*)p;
  v8bf hi = *(const v8bf*)(p + 16);
  return __builtin_shufflevector(lo, hi, 0, 1, 2, 3, 4, 5, 6, 7,
                                 8, 9, 10, 11, 12, 13, 14, 15);
}
// B (32x16 bf16, KxN) staged transposed: sB[n][k]. lane L<16 -> col N=L,
// K = {k0..k0+7, k0+16..k0+23}; lane L>=16 -> col N=L-16, K = {k0+8..k0+15, k0+24..k0+31}.
DEV v16bf load_frag_b(const bf16* s, int n0, int k0, int lane) {
  const bf16* p = s + (n0 + (lane & 15)) * LDSS + k0 + ((lane >> 4) << 3);
  v8bf lo = *(const v8bf*)p;
  v8bf hi = *(const v8bf*)(p + 16);
  return __builtin_shufflevector(lo, hi, 0, 1, 2, 3, 4, 5, 6, 7,
                                 8, 9, 10, 11, 12, 13, 14, 15);
}

DEV v8f wmma_bf16(v16bf a, v16bf b, v8f c) {
  return __builtin_amdgcn_wmma_f32_16x16x32_bf16(false, a, false, b, (short)0,
                                                 c, false, false);
}

// ------------------------- generic C = A * B^T ------------------------------
// A: [M,K] bf16 row-major (lda), B: [N,K] bf16 row-major (ldb)  (i.e. B^T),
// C: [M,N] (ldc) fp32 or bf16.  blockIdx.z batches with element strides.
template <bool OUT_BF16, bool HAS_BIAS>
__global__ __launch_bounds__(256)
void gemm_bt_kernel(const bf16* __restrict__ A, int lda, size_t sAz,
                    const bf16* __restrict__ B, int ldb, size_t sBz,
                    void* __restrict__ Cv, int ldc, size_t sCz,
                    const float* __restrict__ bias, int sBiasZ,
                    int K, float scale) {
#if USE_ASYNC
  __shared__ bf16 sAbuf[2][TM * LDSS];
  __shared__ bf16 sBbuf[2][TN * LDSS];
#else
  __shared__ bf16 sAbuf[1][TM * LDSS];
  __shared__ bf16 sBbuf[1][TN * LDSS];
#endif
  const int tid  = threadIdx.x;
  const int lane = tid & 31;
  const int wave = tid >> 5;               // 8 waves (wave32)
  const int wm   = (wave & 3) * 32;        // 4 waves tile M
  const int wn   = (wave >> 2) * 64;       // 2 waves tile N

  A += blockIdx.z * sAz + (size_t)blockIdx.x * TM * lda;
  B += blockIdx.z * sBz + (size_t)blockIdx.y * TN * ldb;

  v8f acc[2][4] = {};

  const int lrow = tid >> 3;               // 0..31
  const int lcol = (tid & 7) * 8;          // 0..56

  auto compute = [&](const bf16* cA, const bf16* cB) {
#pragma unroll
    for (int kk = 0; kk < TK; kk += 32) {
      v16bf a0 = load_frag_a(cA, wm, kk, lane);
      v16bf a1 = load_frag_a(cA, wm + 16, kk, lane);
#pragma unroll
      for (int ni = 0; ni < 4; ++ni) {
        v16bf b = load_frag_b(cB, wn + ni * 16, kk, lane);
        acc[0][ni] = wmma_bf16(a0, b, acc[0][ni]);
        acc[1][ni] = wmma_bf16(a1, b, acc[1][ni]);
      }
    }
  };

#if USE_ASYNC
  // Double-buffered ASYNCcnt pipeline: 8 async b128 ops per wave per chunk
  // (4 for sA, 4 for sB); wait<=8 retires the oldest chunk (in-order loads).
  auto stage = [&](int buf, int k0) {
#pragma unroll
    for (int r = 0; r < 4; ++r) {
      int row = lrow + r * 32;
      async_cp_b128(&A[(size_t)row * lda + k0 + lcol],
                    &sAbuf[buf][row * LDSS + lcol]);
      async_cp_b128(&B[(size_t)row * ldb + k0 + lcol],
                    &sBbuf[buf][row * LDSS + lcol]);
    }
  };
  stage(0, 0);
  int buf = 0;
  for (int k0 = 0; k0 < K; k0 += TK, buf ^= 1) {
    if (k0 + TK < K) {
      stage(buf ^ 1, k0 + TK);   // safe: buf^1 last read in previous iter (barriered)
      WAIT_ASYNC(8);             // my chunk-k0 copies have landed
    } else {
      WAIT_ASYNC(0);
    }
    __syncthreads();             // everyone's copies have landed
    compute(sAbuf[buf], sBbuf[buf]);
    __syncthreads();             // all reads of buf done before it is rewritten
  }
#else
  for (int k0 = 0; k0 < K; k0 += TK) {
#pragma unroll
    for (int r = 0; r < 4; ++r) {
      int row = lrow + r * 32;
      *(v8bf*)&sAbuf[0][row * LDSS + lcol] =
          *(const v8bf*)&A[(size_t)row * lda + k0 + lcol];
      *(v8bf*)&sBbuf[0][row * LDSS + lcol] =
          *(const v8bf*)&B[(size_t)row * ldb + k0 + lcol];
    }
    __syncthreads();
    compute(sAbuf[0], sBbuf[0]);
    __syncthreads();
  }
#endif

  // Epilogue. C layout: VGPR r -> row = r + 8*(lane>=16), col = lane&15.
  const int rowHi = (lane >> 4) * 8;
  const int colL  = lane & 15;
  const size_t cz = blockIdx.z * sCz;
#pragma unroll
  for (int mi = 0; mi < 2; ++mi)
#pragma unroll
    for (int ni = 0; ni < 4; ++ni)
#pragma unroll
      for (int r = 0; r < 8; ++r) {
        int row = blockIdx.x * TM + wm + mi * 16 + rowHi + r;
        int col = blockIdx.y * TN + wn + ni * 16 + colL;
        float v = acc[mi][ni][r] * scale;
        if (HAS_BIAS) v += bias[(size_t)blockIdx.z * sBiasZ + col];
        if (OUT_BF16)
          ((bf16*)Cv)[cz + (size_t)row * ldc + col] = (bf16)v;
        else
          ((float*)Cv)[cz + (size_t)row * ldc + col] = v;
      }
}

// ------------- per-KEY-column online softmax stats (axis = query s) ---------
__global__ void col_softmax_stats_kernel(const float* __restrict__ scores,
                                         float* __restrict__ mOut,
                                         float* __restrict__ invZOut) {
  const int h   = blockIdx.y;
  const int col = blockIdx.x * blockDim.x + threadIdx.x;   // key index t
  const float* p = scores + (size_t)h * SEQ * SEQ + col;
  float mx = -3.0e38f, z = 0.0f;
  for (int s = 0; s < SEQ; ++s) {
    if (s + 32 < SEQ)
      __builtin_prefetch(p + (size_t)(s + 32) * SEQ, 0, 0);  // 8KB-stride walk
    float x  = p[(size_t)s * SEQ];
    float nm = fmaxf(mx, x);
    z  = z * __expf(mx - nm) + __expf(x - nm);
    mx = nm;
  }
  mOut[h * SEQ + col]    = mx;
  invZOut[h * SEQ + col] = 1.0f / z;
}

// --------- heads = exp(S - m[t]) @ (V[t,:] / Z[t]); out in concat layout -----
__global__ __launch_bounds__(256)
void attn_av_kernel(const float* __restrict__ scores,
                    const bf16* __restrict__ Vh,
                    const float* __restrict__ mBuf,
                    const float* __restrict__ invZ,
                    bf16* __restrict__ outConcat) {
  __shared__ bf16 sA[TM * LDSS];
  __shared__ bf16 sB[TN * LDSS];
  const int h    = blockIdx.z;
  const int tid  = threadIdx.x;
  const int lane = tid & 31;
  const int wave = tid >> 5;
  const int wm   = (wave & 3) * 32;
  const int wn   = (wave >> 2) * 64;

  const float* Sc = scores + (size_t)h * SEQ * SEQ + (size_t)blockIdx.x * TM * SEQ;
  const bf16*  V  = Vh + (size_t)h * SEQ * DK;
  const float* mh = mBuf + h * SEQ;
  const float* zh = invZ + h * SEQ;

  v8f acc[2][4] = {};

  const int arow = tid >> 4;          // 0..15 (A staging row group)
  const int acol = (tid & 15) * 4;    // 0..60 (4 fp32 each)
  const int bcol = (tid & 15) * 8;    // 0..120 (8 bf16 each)

  for (int k0 = 0; k0 < SEQ; k0 += TK) {
    // A tile: exp(score - m[t]) -> bf16
    v4f mt = *(const v4f*)&mh[k0 + acol];
#pragma unroll
    for (int r = 0; r < 8; ++r) {
      int row = arow + r * 16;
      v4f x = *(const v4f*)&Sc[(size_t)row * SEQ + k0 + acol];
      bf16* d = &sA[row * LDSS + acol];
      d[0] = (bf16)__expf(x.x - mt.x);
      d[1] = (bf16)__expf(x.y - mt.y);
      d[2] = (bf16)__expf(x.z - mt.z);
      d[3] = (bf16)__expf(x.w - mt.w);
    }
    // B tile: transpose V rows into sB[n][t], scaled by 1/Z[t]
#pragma unroll
    for (int p = 0; p < 4; ++p) {
      int t = arow + p * 16;          // 0..63 within chunk
      v8bf v = *(const v8bf*)&V[(size_t)(k0 + t) * DK + bcol];
      float iz = zh[k0 + t];
#pragma unroll
      for (int j = 0; j < 8; ++j)
        sB[(bcol + j) * LDSS + t] = (bf16)((float)v[j] * iz);
    }
    __syncthreads();
#pragma unroll
    for (int kk = 0; kk < TK; kk += 32) {
      v16bf a0 = load_frag_a(sA, wm, kk, lane);
      v16bf a1 = load_frag_a(sA, wm + 16, kk, lane);
#pragma unroll
      for (int ni = 0; ni < 4; ++ni) {
        v16bf b = load_frag_b(sB, wn + ni * 16, kk, lane);
        acc[0][ni] = wmma_bf16(a0, b, acc[0][ni]);
        acc[1][ni] = wmma_bf16(a1, b, acc[1][ni]);
      }
    }
    __syncthreads();
  }

  const int rowHi = (lane >> 4) * 8;
  const int colL  = lane & 15;
#pragma unroll
  for (int mi = 0; mi < 2; ++mi)
#pragma unroll
    for (int ni = 0; ni < 4; ++ni)
#pragma unroll
      for (int r = 0; r < 8; ++r) {
        int row = blockIdx.x * TM + wm + mi * 16 + rowHi + r;
        int col = wn + ni * 16 + colL;                 // 0..127
        outConcat[(size_t)row * (NH * DK) + h * DK + col] = (bf16)acc[mi][ni][r];
      }
}

// ---------------------------------------------------------------------------
extern "C" void kernel_launch(void* const* d_in, const int* in_sizes, int n_in,
                              void* d_out, int out_size, void* d_ws, size_t ws_size,
                              hipStream_t stream) {
  const float* query = (const float*)d_in[0];
  const float* key   = (const float*)d_in[1];
  const float* value = (const float*)d_in[2];
  const float* Wq    = (const float*)d_in[3];
  const float* bq    = (const float*)d_in[4];
  const float* Wk    = (const float*)d_in[5];
  const float* bk    = (const float*)d_in[6];
  const float* Wv    = (const float*)d_in[7];
  const float* bv    = (const float*)d_in[8];
  const float* Wo    = (const float*)d_in[9];
  const float* bo    = (const float*)d_in[10];
  float* out = (float*)d_out;

  // -------- workspace carve-up (256B aligned) --------
  char* ws = (char*)d_ws;
  auto alloc = [&](size_t bytes) {
    char* p = ws;
    ws += (bytes + 255) & ~(size_t)255;
    return p;
  };
  const size_t SZ_MAT = (size_t)SEQ * DM * sizeof(bf16);          // 8 MB
  bf16* Xq  = (bf16*)alloc(SZ_MAT);
  bf16* Xk  = (bf16*)alloc(SZ_MAT);
  bf16* Xv  = (bf16*)alloc(SZ_MAT);
  bf16* Wqt = (bf16*)alloc(SZ_MAT);   // [H][DK][DM]
  bf16* Wkt = (bf16*)alloc(SZ_MAT);
  bf16* Wvt = (bf16*)alloc(SZ_MAT);
  bf16* Wot = (bf16*)alloc(SZ_MAT);   // [DM][DM] transposed
  bf16* Qh  = (bf16*)alloc(SZ_MAT);   // [H][SEQ][DK]
  bf16* Kh  = (bf16*)alloc(SZ_MAT);
  bf16* Vh  = (bf16*)alloc(SZ_MAT);
  bf16* Cc  = (bf16*)alloc(SZ_MAT);   // concat heads [SEQ][H*DK]
  float* mB = (float*)alloc((size_t)NH * SEQ * sizeof(float));
  float* zB = (float*)alloc((size_t)NH * SEQ * sizeof(float));
  float* Sf = (float*)alloc((size_t)NH * SEQ * SEQ * sizeof(float)); // 256 MB

  const int nElem = SEQ * DM;                     // 4M
  const int cvtGrid = nElem / (256 * 4);

  // 1) converts
  cvt_bf16_kernel<<<cvtGrid, 256, 0, stream>>>(query, Xq, nElem);
  cvt_bf16_kernel<<<cvtGrid, 256, 0, stream>>>(key,   Xk, nElem);
  cvt_bf16_kernel<<<cvtGrid, 256, 0, stream>>>(value, Xv, nElem);
  {
    const size_t totW = (size_t)NH * DM * DK;     // 4M
    int g = (int)((totW + 255) / 256);
    cvt_transpose_kernel<<<g, 256, 0, stream>>>(Wq, Wqt, DM, DK, NH);
    cvt_transpose_kernel<<<g, 256, 0, stream>>>(Wk, Wkt, DM, DK, NH);
    cvt_transpose_kernel<<<g, 256, 0, stream>>>(Wv, Wvt, DM, DK, NH);
    const size_t totWo = (size_t)DM * DM;
    cvt_transpose_kernel<<<(int)((totWo + 255) / 256), 256, 0, stream>>>(
        Wo, Wot, DM, DM, 1);
  }

  // 2) projections: [SEQ,DM] x [DM,DK] per head -> bf16 [H][SEQ][DK]
  {
    dim3 grid(SEQ / TM, 1, NH);
    gemm_bt_kernel<true, true><<<grid, 256, 0, stream>>>(
        Xq, DM, 0, Wqt, DM, (size_t)DK * DM, Qh, DK, (size_t)SEQ * DK,
        bq, DK, DM, 1.0f);
    gemm_bt_kernel<true, true><<<grid, 256, 0, stream>>>(
        Xk, DM, 0, Wkt, DM, (size_t)DK * DM, Kh, DK, (size_t)SEQ * DK,
        bk, DK, DM, 1.0f);
    gemm_bt_kernel<true, true><<<grid, 256, 0, stream>>>(
        Xv, DM, 0, Wvt, DM, (size_t)DK * DM, Vh, DK, (size_t)SEQ * DK,
        bv, DK, DM, 1.0f);
  }

  // 3) scores = Q K^T / sqrt(DK): fp32 [H][SEQ][SEQ]
  {
    dim3 grid(SEQ / TM, SEQ / TN, NH);
    const float scale = 0.08838834764831845f;     // 1/sqrt(128)
    gemm_bt_kernel<false, false><<<grid, 256, 0, stream>>>(
        Qh, DK, (size_t)SEQ * DK, Kh, DK, (size_t)SEQ * DK,
        Sf, SEQ, (size_t)SEQ * SEQ, nullptr, 0, DK, scale);
  }

  // 4) per-key-column softmax stats (normalize over query axis)
  col_softmax_stats_kernel<<<dim3(SEQ / 256, NH), 256, 0, stream>>>(Sf, mB, zB);

  // 5) heads = exp(S - m) @ (V / Z) -> concat bf16 [SEQ][H*DK]
  attn_av_kernel<<<dim3(SEQ / TM, 1, NH), 256, 0, stream>>>(Sf, Vh, mB, zB, Cc);

  // 6) out = concat @ Wo + bo -> fp32 d_out [SEQ][DM]
  gemm_bt_kernel<false, true><<<dim3(SEQ / TM, DM / TN, 1), 256, 0, stream>>>(
      Cc, DM, 0, Wot, DM, 0, out, DM, 0, bo, 0, DM, 1.0f);
}
